// dismodel_83966610636861
// MI455X (gfx1250) — matrix operand order repaired and checked by reading, not compiled, not run on hardware
//
#include <hip/hip_runtime.h>
#include <hip/hip_bf16.h>
#include <cstdint>

// ---------------------------------------------------------------------------
// Model constants
// ---------------------------------------------------------------------------
#define S_SEG   2048
#define P_TOK   16
#define L_TOK   (S_SEG * P_TOK)      // 32768
#define D_HID   256
#define DP_POS  32
#define BD_BERT 768
#define NH_HEADS 8
#define DK_HEAD 32
#define LAM_MIX 0.8f

typedef _Float16 v16h __attribute__((ext_vector_type(16)));
typedef _Float16 v8h  __attribute__((ext_vector_type(8)));
typedef float    v8f  __attribute__((ext_vector_type(8)));
typedef unsigned int u32x4 __attribute__((ext_vector_type(4)));
typedef int          i32x8 __attribute__((ext_vector_type(8)));
typedef int          i32x4 __attribute__((ext_vector_type(4)));

#if defined(__has_builtin)
#if __has_builtin(__builtin_amdgcn_tensor_load_to_lds) && __has_builtin(__builtin_amdgcn_s_wait_tensorcnt)
#define HAVE_TDM 1
#endif
#endif
#ifndef HAVE_TDM
#define HAVE_TDM 0
#endif

// ---------------------------------------------------------------------------
// Generic WMMA GEMM: C[M,N] = act(A[M,K] @ B[K,N] + bias)
//   Block: 128 threads (4 waves). Tile: 64(M) x 64(N), K step 32.
//   A modes: 0=plain [M,K], 1=embed-concat (bert|posl|posr), 2=conv im2col,
//            3=concat2 (A0 [M,K0] | A1 [M,K-K0])
//   B modes: 0=[K,N], 1=[N,K] (A @ B^T), 2=conv weight [n][c][t]
// ---------------------------------------------------------------------------
struct GemmArgs {
    const float* A0; const float* A1; const float* B; const float* bias; float* C;
    int M, N, K, K0, taps, pad, relu;
    const int* idx0; const int* idx1; const float* emb0; const float* emb1;
};

template <int AM>
__device__ __forceinline__ float fetchA(const GemmArgs& g, int row, int k) {
    if (AM == 0) return g.A0[(long)row * g.K + k];
    if (AM == 1) { // bert(768) | posl_emb[left_pos](32) | posr_emb[right_pos](32)
        if (k < BD_BERT) return g.A0[(long)row * BD_BERT + k];
        if (k < BD_BERT + DP_POS) return g.emb0[(long)g.idx0[row] * DP_POS + (k - BD_BERT)];
        return g.emb1[(long)g.idx1[row] * DP_POS + (k - BD_BERT - DP_POS)];
    }
    if (AM == 2) { // on-the-fly im2col: k = t*256 + c ; row = s*16 + p
        int t = k >> 8, c = k & 255;
        int p = (row & 15) + t - g.pad;
        if ((unsigned)p >= 16u) return 0.0f;
        return g.A0[(long)((row & ~15) + p) * D_HID + c];
    }
    // AM == 3: concat2
    if (k < g.K0) return g.A0[(long)row * g.K0 + k];
    return g.A1[(long)row * (g.K - g.K0) + (k - g.K0)];
}

template <int BM>
__device__ __forceinline__ float fetchB(const GemmArgs& g, int k, int n) {
    if (BM == 0) return g.B[(long)k * g.N + n];
    if (BM == 1) return g.B[(long)n * g.K + k];
    // BM == 2: conv weight [d_out][c][t], k = t*256 + c
    int t = k >> 8, c = k & 255;
    return g.B[(long)n * (D_HID * g.taps) + c * g.taps + t];
}

// Stage 16 contiguous K elements of one row: b128 global loads (fast modes),
// packed f16 conversion, two b128-class LDS stores.
template <int AM>
__device__ __forceinline__ void stageA16(const GemmArgs& g, int row, int k0,
                                         _Float16* __restrict__ dst) {
    _Float16 t[16];
    if (AM == 0) {
        const float* p = g.A0 + (long)row * g.K + k0;
#pragma unroll
        for (int i = 0; i < 4; ++i) {
            float4 v = *(const float4*)(p + 4 * i);
            t[4*i+0] = (_Float16)v.x; t[4*i+1] = (_Float16)v.y;
            t[4*i+2] = (_Float16)v.z; t[4*i+3] = (_Float16)v.w;
        }
    } else {
#pragma unroll
        for (int i = 0; i < 16; ++i) t[i] = (_Float16)fetchA<AM>(g, row, k0 + i);
    }
    v8h a0, a1;
#pragma unroll
    for (int i = 0; i < 8; ++i) { a0[i] = t[i]; a1[i] = t[8 + i]; }
    *(v8h*)(dst) = a0;
    *(v8h*)(dst + 8) = a1;
}

template <int BM>
__device__ __forceinline__ void stageB16(const GemmArgs& g, int n, int k0,
                                         _Float16* __restrict__ dst) {
    _Float16 t[16];
    if (BM == 1) {
        const float* p = g.B + (long)n * g.K + k0;
#pragma unroll
        for (int i = 0; i < 4; ++i) {
            float4 v = *(const float4*)(p + 4 * i);
            t[4*i+0] = (_Float16)v.x; t[4*i+1] = (_Float16)v.y;
            t[4*i+2] = (_Float16)v.z; t[4*i+3] = (_Float16)v.w;
        }
    } else {
#pragma unroll
        for (int i = 0; i < 16; ++i) t[i] = (_Float16)fetchB<BM>(g, k0 + i, n);
    }
    v8h a0, a1;
#pragma unroll
    for (int i = 0; i < 8; ++i) { a0[i] = t[i]; a1[i] = t[8 + i]; }
    *(v8h*)(dst) = a0;
    *(v8h*)(dst + 8) = a1;
}

template <int AM, int BM>
__global__ __launch_bounds__(128)
void gemm_wmma_k(GemmArgs g) {
    __shared__ alignas(32) _Float16 Alds[64 * 32];   // [m_local][k] row-major
    __shared__ alignas(32) _Float16 Blds[64 * 32];   // [n_local][k] (B^T layout)
    const int tid   = threadIdx.x;
    const int lane  = tid & 31;
    const int wave  = tid >> 5;
    const int nbase = blockIdx.x * 64;
    const int mbase = blockIdx.y * 64;

    v8f acc[4];
#pragma unroll
    for (int j = 0; j < 4; ++j)
#pragma unroll
        for (int i = 0; i < 8; ++i) acc[j][i] = 0.0f;

    const int ml = wave * 16 + (lane & 15);
    const int kh = lane >> 4;
    const int srow = tid >> 1;           // 0..63
    const int skseg = (tid & 1) * 16;    // 0 or 16

    for (int kb = 0; kb < g.K; kb += 32) {
        // stage A (64x32) and B^T (64x32) tiles, f32 -> f16, vectorized
        stageA16<AM>(g, mbase + srow, kb + skseg, Alds + srow * 32 + skseg);
        stageB16<BM>(g, nbase + srow, kb + skseg, Blds + srow * 32 + skseg);
        __syncthreads();

        // A fragment (16x32 f16, WMMA A layout): lane m=lane&15 holds
        // K = kh*16..+15 contiguous  -> two contiguous 16B LDS loads
        v8h alo = *(const v8h*)(Alds + ml * 32 + kh * 16);
        v8h ahi = *(const v8h*)(Alds + ml * 32 + kh * 16 + 8);
        v16h a;
#pragma unroll
        for (int i = 0; i < 8; ++i) { a[i] = alo[i]; a[8 + i] = ahi[i]; }

#pragma unroll
        for (int j = 0; j < 4; ++j) {
            // B fragment (32x16): lane n=lane&15 holds K = kh*16..+15 contiguous
            v16h b = *(const v16h*)(Blds + (j * 16 + (lane & 15)) * 32 + kh * 16);
            acc[j] = __builtin_amdgcn_wmma_f32_16x16x32_f16(
                false, a, false, b, (short)0, acc[j], false, false);
        }
        __syncthreads();
    }

    // C layout: lane holds column n=lane&15, rows r + 8*kh
    const int nl = lane & 15;
#pragma unroll
    for (int j = 0; j < 4; ++j) {
        int col = nbase + j * 16 + nl;
        float bv = g.bias ? g.bias[col] : 0.0f;
#pragma unroll
        for (int r = 0; r < 8; ++r) {
            int row = mbase + wave * 16 + 8 * kh + r;
            float v = acc[j][r] + bv;
            if (g.relu) v = fmaxf(v, 0.0f);
            g.C[(long)row * g.N + col] = v;
        }
    }
}

// ---------------------------------------------------------------------------
// Dual-stream flash attention (no 1/sqrt(dk) scaling, per reference).
// One wave per block; grid = (S/16 q-tiles, NH heads).
// ctx = LAM * softmax(Q K^T) V + (1-LAM) * softmax(QE KE^T) V
// ---------------------------------------------------------------------------
__global__ __launch_bounds__(32)
void flash_attn_k(const float* __restrict__ Q, const float* __restrict__ Kk,
                  const float* __restrict__ V, const float* __restrict__ QE,
                  const float* __restrict__ KE, float* __restrict__ CTX) {
    __shared__ alignas(32) _Float16 Px[16 * 32];  // prob tile, x-stream
    __shared__ alignas(32) _Float16 Pe[16 * 32];  // prob tile, e-stream
    __shared__ alignas(32) _Float16 Vt[32 * 32];  // V^T: [dim][key]

    const int lane  = threadIdx.x;
    const int h     = blockIdx.y;
    const int q0    = blockIdx.x * 16;
    const int m     = lane & 15;
    const int kh    = lane >> 4;
    const int dbase = h * DK_HEAD;

    // Q fragments (both streams), A layout
    v16h qf, qef;
#pragma unroll
    for (int i = 0; i < 8; ++i) {
        long base = (long)(q0 + m) * D_HID + dbase;
        qf[i]      = (_Float16)Q[base + kh * 16 + i];
        qf[8 + i]  = (_Float16)Q[base + kh * 16 + 8 + i];
        qef[i]     = (_Float16)QE[base + kh * 16 + i];
        qef[8 + i] = (_Float16)QE[base + kh * 16 + 8 + i];
    }

    v8f cx0, cx1, ce0, ce1;
    float Mx[8], Sx[8], Me[8], Se[8];
#pragma unroll
    for (int r = 0; r < 8; ++r) {
        cx0[r] = cx1[r] = ce0[r] = ce1[r] = 0.0f;
        Mx[r] = Me[r] = -3.0e38f; Sx[r] = Se[r] = 0.0f;
    }

    for (int kb = 0; kb < S_SEG; kb += 32) {
        // key fragments (B layout): lane n = key, dims kh*16..+15 contiguous
        v16h kf[2], kef[2];
#pragma unroll
        for (int j = 0; j < 2; ++j) {
            long krow = (long)(kb + j * 16 + m) * D_HID + dbase + kh * 16;
#pragma unroll
            for (int i = 0; i < 16; ++i) {
                kf[j][i]  = (_Float16)Kk[krow + i];
                kef[j][i] = (_Float16)KE[krow + i];
            }
        }
        v8f z;
#pragma unroll
        for (int r = 0; r < 8; ++r) z[r] = 0.0f;

        v8f lx0 = __builtin_amdgcn_wmma_f32_16x16x32_f16(false, qf,  false, kf[0],  (short)0, z, false, false);
        v8f lx1 = __builtin_amdgcn_wmma_f32_16x16x32_f16(false, qf,  false, kf[1],  (short)0, z, false, false);
        v8f le0 = __builtin_amdgcn_wmma_f32_16x16x32_f16(false, qef, false, kef[0], (short)0, z, false, false);
        v8f le1 = __builtin_amdgcn_wmma_f32_16x16x32_f16(false, qef, false, kef[1], (short)0, z, false, false);

        float pX0[8], pX1[8], pE0[8], pE1[8];
#pragma unroll
        for (int r = 0; r < 8; ++r) {
            // --- x stream online softmax, row = r + 8*kh (lanes kh*16..+15)
            float mx = fmaxf(lx0[r], lx1[r]);
#pragma unroll
            for (int off = 8; off >= 1; off >>= 1) mx = fmaxf(mx, __shfl_xor(mx, off, 16));
            float Mn = fmaxf(Mx[r], mx);
            float al = __expf(Mx[r] - Mn);
            Mx[r] = Mn; cx0[r] *= al; cx1[r] *= al; Sx[r] *= al;
            pX0[r] = __expf(lx0[r] - Mn); pX1[r] = __expf(lx1[r] - Mn);
            float rs = pX0[r] + pX1[r];
#pragma unroll
            for (int off = 8; off >= 1; off >>= 1) rs += __shfl_xor(rs, off, 16);
            Sx[r] += rs;
            // --- e stream
            float me = fmaxf(le0[r], le1[r]);
#pragma unroll
            for (int off = 8; off >= 1; off >>= 1) me = fmaxf(me, __shfl_xor(me, off, 16));
            float Mne = fmaxf(Me[r], me);
            float ale = __expf(Me[r] - Mne);
            Me[r] = Mne; ce0[r] *= ale; ce1[r] *= ale; Se[r] *= ale;
            pE0[r] = __expf(le0[r] - Mne); pE1[r] = __expf(le1[r] - Mne);
            float rse = pE0[r] + pE1[r];
#pragma unroll
            for (int off = 8; off >= 1; off >>= 1) rse += __shfl_xor(rse, off, 16);
            Se[r] += rse;
        }

        // P tiles: C layout -> LDS row-major [16 q][32 key]
#pragma unroll
        for (int r = 0; r < 8; ++r) {
            int row = r + 8 * kh;
            Px[row * 32 + m]      = (_Float16)pX0[r];
            Px[row * 32 + 16 + m] = (_Float16)pX1[r];
            Pe[row * 32 + m]      = (_Float16)pE0[r];
            Pe[row * 32 + 16 + m] = (_Float16)pE1[r];
        }
        // V^T tile: lane loads one key row (coalesced), scatters to [dim][key]
#pragma unroll
        for (int dd = 0; dd < 32; ++dd)
            Vt[dd * 32 + lane] = (_Float16)V[(long)(kb + lane) * D_HID + dbase + dd];
        __syncthreads();

        // reload P as A fragments, V^T as B fragments
        v16h pxf, pef;
#pragma unroll
        for (int i = 0; i < 8; ++i) {
            pxf[i]     = Px[m * 32 + kh * 16 + i];
            pxf[8 + i] = Px[m * 32 + kh * 16 + 8 + i];
            pef[i]     = Pe[m * 32 + kh * 16 + i];
            pef[8 + i] = Pe[m * 32 + kh * 16 + 8 + i];
        }
        v16h vf0, vf1;
#pragma unroll
        for (int i = 0; i < 16; ++i) {
            vf0[i] = Vt[(m) * 32 + kh * 16 + i];
            vf1[i] = Vt[(16 + m) * 32 + kh * 16 + i];
        }
        cx0 = __builtin_amdgcn_wmma_f32_16x16x32_f16(false, pxf, false, vf0, (short)0, cx0, false, false);
        cx1 = __builtin_amdgcn_wmma_f32_16x16x32_f16(false, pxf, false, vf1, (short)0, cx1, false, false);
        ce0 = __builtin_amdgcn_wmma_f32_16x16x32_f16(false, pef, false, vf0, (short)0, ce0, false, false);
        ce1 = __builtin_amdgcn_wmma_f32_16x16x32_f16(false, pef, false, vf1, (short)0, ce1, false, false);
        __syncthreads();
    }

#pragma unroll
    for (int r = 0; r < 8; ++r) {
        int row = q0 + 8 * kh + r;
        float o0 = LAM_MIX * cx0[r] / Sx[r] + (1.0f - LAM_MIX) * ce0[r] / Se[r];
        float o1 = LAM_MIX * cx1[r] / Sx[r] + (1.0f - LAM_MIX) * ce1[r] / Se[r];
        CTX[(long)row * D_HID + dbase + m]      = o0;
        CTX[(long)row * D_HID + dbase + 16 + m] = o1;
    }
}

// ---------------------------------------------------------------------------
// TDM: DMA one contiguous row of `nfloats` f32 from global into LDS.
// Builds a minimal 2D tensor descriptor (D#): tile = tensor = [nfloats x 1].
// 6-arg builtin form: (g0 u32x4, g1 i32x8, g2 i32x4, g3 i32x4, i32x8, cpol)
// ---------------------------------------------------------------------------
#if HAVE_TDM
__device__ __forceinline__ void tdm_load_row(const float* gsrc, float* lds_dst, int nfloats) {
    unsigned long long ga = (unsigned long long)(uintptr_t)gsrc;
    unsigned lds = (unsigned)(uintptr_t)lds_dst;   // low 32 bits = LDS offset
    u32x4 g0;
    g0[0] = 1u;                                        // count=1, user descriptor
    g0[1] = lds;                                       // lds_addr (bytes)
    g0[2] = (unsigned)(ga & 0xFFFFFFFFu);              // global_addr[31:0]
    g0[3] = (unsigned)((ga >> 32) & 0x01FFFFFFu)       // global_addr[56:32]
          | (2u << 30);                                // type = 2 ("image")
    i32x8 g1;
    g1[0] = (2 << 16);                                 // data_size = 2 (4 bytes)
    g1[1] = (nfloats & 0xFFFF) << 16;                  // tensor_dim0[15:0] @ bits 63:48
    g1[2] = ((nfloats >> 16) & 0xFFFF)                 // tensor_dim0[31:16]
          | (1 << 16);                                 // tensor_dim1 = 1 (low bits)
    g1[3] = (nfloats & 0xFFFF) << 16;                  // tile_dim0 @ bits 127:112
    g1[4] = 1;                                         // tile_dim1 = 1
    g1[5] = nfloats;                                   // tensor_dim0_stride[31:0]
    g1[6] = 0;
    g1[7] = 0;
    i32x4 g2; g2[0] = 0; g2[1] = 0; g2[2] = 0; g2[3] = 0;
    i32x4 g3; g3[0] = 0; g3[1] = 0; g3[2] = 0; g3[3] = 0;
    i32x8 g4;
#pragma unroll
    for (int i = 0; i < 8; ++i) g4[i] = 0;
    __builtin_amdgcn_tensor_load_to_lds(g0, g1, g2, g3, g4, 0);
}
#endif

// ---------------------------------------------------------------------------
// GRU recurrence: persistent block per direction. Input gates gi precomputed
// by WMMA GEMMs; only h @ W_hh^T is sequential. The per-step gate row
// gi[s,0:768] is DMA'd into LDS by the Tensor Data Mover, overlapped with the
// 768x256 matvec, and published by the existing block barrier.
// ---------------------------------------------------------------------------
struct GruArgs {
    const float* gi0; const float* gi1;
    const float* whh0; const float* whh1;
    const float* bhh0; const float* bhh1;
    float* out0; float* out1;
    int outStride; int off0; int off1;
};

__global__ __launch_bounds__(768)
void gru_dir_k(GruArgs a) {
    const int dir = blockIdx.x;
    const float* gi  = dir ? a.gi1  : a.gi0;
    const float* whh = dir ? a.whh1 : a.whh0;
    const float* bhh = dir ? a.bhh1 : a.bhh0;
    float* out = dir ? a.out1 : a.out0;
    const int off = dir ? a.off1 : a.off0;

    __shared__ alignas(16) float h[256];
    __shared__ float gh[768];
    __shared__ alignas(16) float gih[768];
    const int tid = threadIdx.x;
    if (tid < 256) h[tid] = 0.0f;
    __syncthreads();

    const float* wr = whh + (long)tid * 256;
    const float bb = bhh[tid];

    for (int step = 0; step < S_SEG; ++step) {
        int s = dir ? (S_SEG - 1 - step) : step;
#if HAVE_TDM
        if (tid < 32) tdm_load_row(gi + (long)s * 768, gih, 768);  // async DMA
#endif
        float acc = bb;
#pragma unroll 4
        for (int c = 0; c < 256; c += 4) {
            float4 hv = *(const float4*)(h + c);
            float4 wv = *(const float4*)(wr + c);
            acc += hv.x * wv.x + hv.y * wv.y + hv.z * wv.z + hv.w * wv.w;
        }
#if HAVE_TDM
        if (tid < 32) __builtin_amdgcn_s_wait_tensorcnt(0);
#endif
        gh[tid] = acc;
        __syncthreads();   // publishes gh and the TDM-written gih

        if (tid < 256) {
#if HAVE_TDM
            float ir = gih[tid], iz = gih[256 + tid], inn = gih[512 + tid];
#else
            const float* grow = gi + (long)s * 768;
            float ir = grow[tid], iz = grow[256 + tid], inn = grow[512 + tid];
#endif
            float r = 1.0f / (1.0f + __expf(-(ir + gh[tid])));
            float z = 1.0f / (1.0f + __expf(-(iz + gh[256 + tid])));
            float n = tanhf(inn + r * gh[512 + tid]);
            float hn = (1.0f - z) * n + z * h[tid];
            out[(long)s * a.outStride + off + tid] = hn;
            h[tid] = hn;
        }
        __syncthreads();
    }
}

// ---------------------------------------------------------------------------
// Small helper kernels
// ---------------------------------------------------------------------------
__global__ __launch_bounds__(256) void maxpool_k(const float* __restrict__ H2, float* __restrict__ SEG) {
    int s = blockIdx.x, c = threadIdx.x;
    float m = H2[(long)(s * 16) * D_HID + c];
#pragma unroll
    for (int p = 1; p < 16; ++p) m = fmaxf(m, H2[(long)(s * 16 + p) * D_HID + c]);
    SEG[(long)s * D_HID + c] = m;
}

__global__ __launch_bounds__(256) void exgather_k(const float* __restrict__ entemb, const int* __restrict__ et, float* __restrict__ EX) {
    int s = blockIdx.x, c = threadIdx.x;
    EX[(long)s * D_HID + c] = entemb[(long)et[s] * D_HID + c];
}

__global__ __launch_bounds__(256) void fuse_k(const float* GP, const float* GIN, const float* SEG, float* FUS, int n) {
    int i = blockIdx.x * 256 + threadIdx.x;
    if (i < n) { float g = GP[i]; FUS[i] = g * GIN[i] + (1.0f - g) * SEG[i]; }
}

__global__ __launch_bounds__(256) void add_k(const float* A, const float* B, float* C, int n) {
    int i = blockIdx.x * 256 + threadIdx.x;
    if (i < n) C[i] = A[i] + B[i];
}

// t[c] = sum_d scorew1[d][c]*symy[d] ; T[256] = sum_d symy[d]*scorew2[256+d]
__global__ __launch_bounds__(256)
void score_sym_k(const float* __restrict__ modely, const float* __restrict__ scorew1,
                 const float* __restrict__ scorew2, const int* __restrict__ symp, float* __restrict__ T) {
    __shared__ float red[256];
    int c = threadIdx.x;
    int sym = symp[0];
    const float* symy = modely + (long)sym * D_HID;
    float acc = 0.0f;
    for (int d = 0; d < 256; ++d) acc += scorew1[(long)d * D_HID + c] * symy[d];
    T[c] = acc;
    red[c] = symy[c] * scorew2[256 + c];
    __syncthreads();
    for (int off = 128; off; off >>= 1) { if (c < off) red[c] += red[c + off]; __syncthreads(); }
    if (c == 0) T[256] = red[0];
}

__global__ __launch_bounds__(256)
void score_k(const float* __restrict__ modely, const float* __restrict__ T,
             const float* __restrict__ scorew2, const float* __restrict__ scorec, float* __restrict__ outp) {
    int s = blockIdx.x * 256 + threadIdx.x;
    const float* row = modely + (long)s * D_HID;
    float q = 0.0f, l = 0.0f;
    for (int c = 0; c < 256; ++c) { q += row[c] * T[c]; l += row[c] * scorew2[c]; }
    float v = q + l + T[256] + scorec[0];
    outp[s] = 1.0f / (1.0f + __expf(-v));
}

// ---------------------------------------------------------------------------
// Host orchestration
// ---------------------------------------------------------------------------
extern "C" void kernel_launch(void* const* d_in, const int* in_sizes, int n_in,
                              void* d_out, int out_size, void* d_ws, size_t ws_size,
                              hipStream_t stream) {
    (void)in_sizes; (void)n_in; (void)out_size; (void)ws_size;
    const float* berthid  = (const float*)d_in[0];
    const int*   left_pos = (const int*)d_in[1];
    const int*   right_pos= (const int*)d_in[2];
    const int*   ent_type = (const int*)d_in[3];
    const int*   sym_idx  = (const int*)d_in[4];
    const float* posl_emb = (const float*)d_in[5];
    const float* posr_emb = (const float*)d_in[6];
    const float* space_w  = (const float*)d_in[7];
    const float* space_b  = (const float*)d_in[8];
    const float* entemb   = (const float*)d_in[9];
    const float* conv1_w  = (const float*)d_in[10];
    const float* conv1_b  = (const float*)d_in[11];
    const float* conv2_w  = (const float*)d_in[12];
    const float* conv2_b  = (const float*)d_in[13];
    const float* wq  = (const float*)d_in[14];
    const float* wk  = (const float*)d_in[15];
    const float* wv  = (const float*)d_in[16];
    const float* ewq = (const float*)d_in[17];
    const float* ewk = (const float*)d_in[18];
    const float* out_w   = (const float*)d_in[19];
    const float* scorew1 = (const float*)d_in[20];
    const float* bq  = (const float*)d_in[21];
    const float* bk  = (const float*)d_in[22];
    const float* bvv = (const float*)d_in[23];
    const float* ebq = (const float*)d_in[24];
    const float* ebk = (const float*)d_in[25];
    const float* out_b  = (const float*)d_in[26];
    const float* gate_w = (const float*)d_in[27];
    const float* gate_b = (const float*)d_in[28];
    const float* w_ih0  = (const float*)d_in[29];  // [2,768,256]
    const float* w_hh0  = (const float*)d_in[30];  // [2,768,256]
    const float* b_ih0  = (const float*)d_in[31];  // [2,768]
    const float* b_hh0  = (const float*)d_in[32];
    const float* w_ih1  = (const float*)d_in[33];  // [2,768,512]
    const float* w_hh1  = (const float*)d_in[34];  // [2,768,256]
    const float* b_ih1  = (const float*)d_in[35];
    const float* b_hh1  = (const float*)d_in[36];
    const float* scorew2 = (const float*)d_in[37];
    const float* scorec  = (const float*)d_in[38];

    float* outp   = (float*)d_out;
    float* modely = outp;                               // [S, 256]
    float* scorep = outp + (size_t)S_SEG * D_HID;       // [S]

    char* wp = (char*)d_ws;
    auto alloc = [&](size_t nfloats) -> float* {
        float* p = (float*)wp;
        wp += ((nfloats * sizeof(float)) + 255) & ~(size_t)255;
        return p;
    };
    float* Xh   = alloc((size_t)L_TOK * D_HID);
    float* H1   = alloc((size_t)L_TOK * D_HID);
    float* H2   = alloc((size_t)L_TOK * D_HID);
    float* SEG  = alloc((size_t)S_SEG * D_HID);
    float* EX   = alloc((size_t)S_SEG * D_HID);
    float* Qb   = alloc((size_t)S_SEG * D_HID);
    float* Kb   = alloc((size_t)S_SEG * D_HID);
    float* Vb   = alloc((size_t)S_SEG * D_HID);
    float* QEb  = alloc((size_t)S_SEG * D_HID);
    float* KEb  = alloc((size_t)S_SEG * D_HID);
    float* CTX  = alloc((size_t)S_SEG * D_HID);
    float* GIN  = alloc((size_t)S_SEG * D_HID);
    float* GP   = alloc((size_t)S_SEG * D_HID);
    float* FUS  = alloc((size_t)S_SEG * D_HID);
    float* GI0F = alloc((size_t)S_SEG * 768);
    float* GI0B = alloc((size_t)S_SEG * 768);
    float* L0   = alloc((size_t)S_SEG * 512);
    float* GI1F = alloc((size_t)S_SEG * 768);
    float* GI1B = alloc((size_t)S_SEG * 768);
    float* F1   = alloc((size_t)S_SEG * D_HID);
    float* B1   = alloc((size_t)S_SEG * D_HID);
    float* Tbuf = alloc(257);

    const dim3 blk(128);
    auto mk = [](const float* A0, const float* A1, const float* B, const float* bias, float* C,
                 int M, int N, int K, int K0, int taps, int pad, int relu,
                 const int* i0, const int* i1, const float* e0, const float* e1) {
        GemmArgs g{A0, A1, B, bias, C, M, N, K, K0, taps, pad, relu, i0, i1, e0, e1};
        return g;
    };

    // 1) space projection: gather-concat [L,832] @ space_w + b -> Xh
    gemm_wmma_k<1, 0><<<dim3(D_HID / 64, L_TOK / 64), blk, 0, stream>>>(
        mk(berthid, nullptr, space_w, space_b, Xh, L_TOK, D_HID, 832, 0, 0, 0, 0,
           left_pos, right_pos, posl_emb, posr_emb));
    // 2) conv1 (k=3, pad=1) + ReLU -> H1
    gemm_wmma_k<2, 2><<<dim3(D_HID / 64, L_TOK / 64), blk, 0, stream>>>(
        mk(Xh, nullptr, conv1_w, conv1_b, H1, L_TOK, D_HID, 768, 0, 3, 1, 1,
           nullptr, nullptr, nullptr, nullptr));
    // 3) conv2 (k=5, pad=2) + ReLU -> H2
    gemm_wmma_k<2, 2><<<dim3(D_HID / 64, L_TOK / 64), blk, 0, stream>>>(
        mk(H1, nullptr, conv2_w, conv2_b, H2, L_TOK, D_HID, 1280, 0, 5, 2, 1,
           nullptr, nullptr, nullptr, nullptr));
    // 4) per-segment maxpool, 5) entity embedding gather
    maxpool_k<<<S_SEG, 256, 0, stream>>>(H2, SEG);
    exgather_k<<<S_SEG, 256, 0, stream>>>(entemb, ent_type, EX);
    // 6) q/k/v/qe/ke projections
    const dim3 pg(D_HID / 64, S_SEG / 64);
    gemm_wmma_k<0, 0><<<pg, blk, 0, stream>>>(mk(SEG, nullptr, wq,  bq,  Qb,  S_SEG, D_HID, D_HID, 0, 0, 0, 0, nullptr, nullptr, nullptr, nullptr));
    gemm_wmma_k<0, 0><<<pg, blk, 0, stream>>>(mk(SEG, nullptr, wk,  bk,  Kb,  S_SEG, D_HID, D_HID, 0, 0, 0, 0, nullptr, nullptr, nullptr, nullptr));
    gemm_wmma_k<0, 0><<<pg, blk, 0, stream>>>(mk(SEG, nullptr, wv,  bvv, Vb,  S_SEG, D_HID, D_HID, 0, 0, 0, 0, nullptr, nullptr, nullptr, nullptr));
    gemm_wmma_k<0, 0><<<pg, blk, 0, stream>>>(mk(EX,  nullptr, ewq, ebq, QEb, S_SEG, D_HID, D_HID, 0, 0, 0, 0, nullptr, nullptr, nullptr, nullptr));
    gemm_wmma_k<0, 0><<<pg, blk, 0, stream>>>(mk(EX,  nullptr, ewk, ebk, KEb, S_SEG, D_HID, D_HID, 0, 0, 0, 0, nullptr, nullptr, nullptr, nullptr));
    // 7) dual-stream flash attention -> CTX
    flash_attn_k<<<dim3(S_SEG / 16, NH_HEADS), 32, 0, stream>>>(Qb, Kb, Vb, QEb, KEb, CTX);
    // 8) g_in = ctx @ out_w + out_b
    gemm_wmma_k<0, 0><<<pg, blk, 0, stream>>>(mk(CTX, nullptr, out_w, out_b, GIN, S_SEG, D_HID, D_HID, 0, 0, 0, 0, nullptr, nullptr, nullptr, nullptr));
    // 9) gp = [g_in|seg] @ gate_w + gate_b (raw, no sigmoid)
    gemm_wmma_k<3, 0><<<pg, blk, 0, stream>>>(mk(GIN, SEG, gate_w, gate_b, GP, S_SEG, D_HID, 512, 256, 0, 0, 0, nullptr, nullptr, nullptr, nullptr));
    // 10) fused = gp*g_in + (1-gp)*seg
    fuse_k<<<(S_SEG * D_HID) / 256, 256, 0, stream>>>(GP, GIN, SEG, FUS, S_SEG * D_HID);
    // 11) GRU layer0 input gates (batched, B^T mode), then recurrence
    const dim3 gg(768 / 64, S_SEG / 64);
    gemm_wmma_k<0, 1><<<gg, blk, 0, stream>>>(mk(FUS, nullptr, w_ih0,             b_ih0,       GI0F, S_SEG, 768, 256, 0, 0, 0, 0, nullptr, nullptr, nullptr, nullptr));
    gemm_wmma_k<0, 1><<<gg, blk, 0, stream>>>(mk(FUS, nullptr, w_ih0 + 768 * 256, b_ih0 + 768, GI0B, S_SEG, 768, 256, 0, 0, 0, 0, nullptr, nullptr, nullptr, nullptr));
    {
        GruArgs a{GI0F, GI0B, w_hh0, w_hh0 + 768 * 256, b_hh0, b_hh0 + 768, L0, L0, 512, 0, 256};
        gru_dir_k<<<2, 768, 0, stream>>>(a);
    }
    // 12) GRU layer1
    gemm_wmma_k<0, 1><<<gg, blk, 0, stream>>>(mk(L0, nullptr, w_ih1,             b_ih1,       GI1F, S_SEG, 768, 512, 0, 0, 0, 0, nullptr, nullptr, nullptr, nullptr));
    gemm_wmma_k<0, 1><<<gg, blk, 0, stream>>>(mk(L0, nullptr, w_ih1 + 768 * 512, b_ih1 + 768, GI1B, S_SEG, 768, 512, 0, 0, 0, 0, nullptr, nullptr, nullptr, nullptr));
    {
        GruArgs a{GI1F, GI1B, w_hh1, w_hh1 + 768 * 256, b_hh1, b_hh1 + 768, F1, B1, 256, 0, 0};
        gru_dir_k<<<2, 768, 0, stream>>>(a);
    }
    // 13) modely = f1 + b1 (written straight into d_out)
    add_k<<<(S_SEG * D_HID) / 256, 256, 0, stream>>>(F1, B1, modely, S_SEG * D_HID);
    // 14) scoring
    score_sym_k<<<1, 256, 0, stream>>>(modely, scorew1, scorew2, sym_idx, Tbuf);
    score_k<<<S_SEG / 256, 256, 0, stream>>>(modely, Tbuf, scorew2, scorec, scorep);
}
